// BiFlowNFLOB_76081050681624
// MI455X (gfx1250) — matrix operand order repaired and checked by reading, not compiled, not run on hardware
//
#include <hip/hip_runtime.h>
#include <hip/hip_bf16.h>

typedef __attribute__((ext_vector_type(16))) __bf16 v16bf;
typedef __attribute__((ext_vector_type(8)))  __bf16 v8bf;
typedef __attribute__((ext_vector_type(8)))  float  v8f;

union U16bf { v16bf v; v8bf h[2]; };

#define LAYERS     6
#define DIM        40
#define ADIM       20
#define HWID       128
#define DIN        276
#define DINP       288
#define BTOT       131072
// per-layer bf16 weight block in d_ws: W1T[128][288] | W2T[128][128] | W3T[128][128] | WstT[64][128]
#define W1T_N      (128*288)     // 36864
#define W2T_OFF    36864
#define W3T_OFF    53248
#define WST_OFF    69632
#define LAYER_WS   77824
#define WTAIL_N    (LAYER_WS - W2T_OFF)   // 40960 = W2T|W3T|WstT merged

// ---------------- weight prep: fp32 -> bf16, transposed + padded ----------------
__global__ void prep_weights(const float* __restrict__ W1, const float* __restrict__ W2,
                             const float* __restrict__ W3, const float* __restrict__ Ws,
                             const float* __restrict__ Wt, __bf16* __restrict__ ws) {
    int idx = blockIdx.x * blockDim.x + threadIdx.x;
    if (idx >= LAYERS * LAYER_WS) return;
    int l = idx / LAYER_WS;
    int e = idx % LAYER_WS;
    float v;
    if (e < W1T_N) {
        int n = e / DINP, k = e % DINP;
        v = (k < DIN) ? W1[((size_t)l * DIN + k) * HWID + n] : 0.0f;
    } else if (e < W3T_OFF) {
        int e2 = e - W2T_OFF; int n = e2 / HWID, k = e2 % HWID;
        v = W2[((size_t)l * HWID + k) * HWID + n];
    } else if (e < WST_OFF) {
        int e2 = e - W3T_OFF; int n = e2 / HWID, k = e2 % HWID;
        v = W3[((size_t)l * HWID + k) * HWID + n];
    } else {
        int e2 = e - WST_OFF; int n = e2 / HWID, k = e2 % HWID;
        if (n < 20)       v = Ws[((size_t)l * HWID + k) * ADIM + n];
        else if (n < 32)  v = 0.0f;
        else if (n < 52)  v = Wt[((size_t)l * HWID + k) * ADIM + (n - 32)];
        else              v = 0.0f;
    }
    ws[idx] = (__bf16)v;
}

// ---------------- cooperative LDS staging (b128 copies) ----------------
__device__ inline void stage(__bf16* dst, const __bf16* src, int nelem) {
    uint4* d = (uint4*)dst;
    const uint4* s = (const uint4*)src;
    int n16 = nelem >> 3;
    for (int i = threadIdx.x; i < n16; i += blockDim.x) d[i] = s[i];
}

// ---------------- wave-level 16xN GEMM on WMMA bf16 ----------------
// A: [16][K] bf16 in LDS (row stride SA) -- hoisted into registers once.
// Bw: [N][K] bf16 in LDS (transposed weights, row stride SB).
// C = A*B^T (+bias), optional SiLU, written bf16 (stride SOUT) or f32.
// B fragments are software-pipelined: B[kt+1] is in flight while WMMA[kt] executes.
template<int SA, int SB, int SOUT, int KT, int NT, bool SILU, bool F32OUT, bool BIAS>
__device__ inline void wave_gemm(const __bf16* __restrict__ A, const __bf16* __restrict__ Bw,
                                 const float* __restrict__ bias,
                                 __bf16* __restrict__ outB, float* __restrict__ outF, int lane) {
    const int col = lane & 15;   // C-fragment column == A row / B row for frag loads
    const int sel = lane >> 4;
    const __bf16* arow = A + col * SA;

    // preload all biases so their global loads overlap the A-fragment LDS loads
    float bvals[NT];
    if constexpr (BIAS) {
#pragma unroll
        for (int nt = 0; nt < NT; ++nt) bvals[nt] = bias[nt * 16 + col];
    }

    // A fragments are invariant across N-tiles: load once.
    // A 16x32 frag: lanes 0-15 hold K = 0..7,16..23 ; lanes 16-31 hold K = 8..15,24..31
    v8bf alo[KT], ahi[KT];
#pragma unroll
    for (int kt = 0; kt < KT; ++kt) {
        alo[kt] = *(const v8bf*)(arow + kt * 32 + sel * 8);
        ahi[kt] = *(const v8bf*)(arow + kt * 32 + 16 + sel * 8);
    }

#pragma unroll
    for (int nt = 0; nt < NT; ++nt) {
        float bval = 0.0f;
        if constexpr (BIAS) bval = bvals[nt];
        v8f acc = {bval, bval, bval, bval, bval, bval, bval, bval};
        const __bf16* brow = Bw + (nt * 16 + col) * SB;

        // prologue: first B fragment
        // B 32x16 frag: lanes 0-15 hold K = 0..7,16..23 ; lanes 16-31 hold K = 8..15,24..31 (N = lane&15)
        U16bf bcur;
        bcur.h[0] = *(const v8bf*)(brow + sel * 8);
        bcur.h[1] = *(const v8bf*)(brow + 16 + sel * 8);
#pragma unroll
        for (int kt = 0; kt < KT; ++kt) {
            U16bf bnext;
            if (kt + 1 < KT) {
                bnext.h[0] = *(const v8bf*)(brow + (kt + 1) * 32 + sel * 8);
                bnext.h[1] = *(const v8bf*)(brow + (kt + 1) * 32 + 16 + sel * 8);
            }
            U16bf a;
            a.h[0] = alo[kt];
            a.h[1] = ahi[kt];
            acc = __builtin_amdgcn_wmma_f32_16x16x32_bf16(false, a.v, false, bcur.v,
                                                          (short)0, acc, false, false);
            if (kt + 1 < KT) bcur = bnext;
        }

        // batched epilogue: 8 independent transcendental chains interleave
        float xs[8];
#pragma unroll
        for (int v = 0; v < 8; ++v) xs[v] = acc[v];
        if constexpr (SILU) {
            float ex[8];
#pragma unroll
            for (int v = 0; v < 8; ++v) ex[v] = __expf(-xs[v]);
#pragma unroll
            for (int v = 0; v < 8; ++v) ex[v] = __builtin_amdgcn_rcpf(1.0f + ex[v]);
#pragma unroll
            for (int v = 0; v < 8; ++v) xs[v] = xs[v] * ex[v];
        }
#pragma unroll
        for (int v = 0; v < 8; ++v) {
            const int m = v + sel * 8;
            if (F32OUT) outF[m * SOUT + nt * 16 + col] = xs[v];
            else        outB[m * SOUT + nt * 16 + col] = (__bf16)xs[v];
        }
    }
}

// ---------------- fused 6-layer coupling flow ----------------
__global__ void __launch_bounds__(128, 2)
flow_kernel(const float* __restrict__ x, const float* __restrict__ ctx,
            const float* __restrict__ te,
            const float* __restrict__ b1, const float* __restrict__ b2,
            const float* __restrict__ b3, const float* __restrict__ bs,
            const float* __restrict__ bt,
            const int* __restrict__ perm, const int* __restrict__ idx_a,
            const int* __restrict__ idx_b,
            const __bf16* __restrict__ ws, float* __restrict__ out) {
    __shared__ __align__(16) __bf16 sW[40960];        // staged weights (W1T or W2T|W3T|WstT)  80 KB
    __shared__ __align__(16) __bf16 sA[4 * 16 * 288]; // concat input [zA|ctx|tE|pad]          36 KB
    __shared__ __align__(16) __bf16 sB[4 * 16 * 128]; // act ping                              16 KB
    __shared__ __align__(16) __bf16 sC[4 * 16 * 128]; // act pong / f32 s,t                    16 KB
    __shared__ __align__(16) float  sZ[4 * 16 * 40];  // per-row flow state                    10 KB

    const int lane = threadIdx.x & 31;
    const int wave = threadIdx.x >> 5;
    const int g0   = blockIdx.x * 64 + wave * 16;     // first global row of this wave's tile

    __bf16* ha = sA + wave * 16 * 288;
    __bf16* hb = sB + wave * 16 * 128;
    __bf16* hc = sC + wave * 16 * 128;
    float*  stw = ((float*)sC) + wave * 16 * 64;      // s|t f32 view, aliases hc region
    float*  zt  = sZ + wave * 16 * 40;

    // ---- one-time loads: x into z-state, ctx & t_e bf16 into concat buffer ----
    for (int i = lane; i < 16 * 40; i += 32) {
        int r = i / 40, c = i % 40;
        zt[r * 40 + c] = x[(size_t)(g0 + r) * 40 + c];
    }
    for (int i = lane; i < 16 * 128; i += 32) {
        int r = i >> 7, c = i & 127;
        ha[r * 288 + 20 + c]  = (__bf16)ctx[(size_t)(g0 + r) * 128 + c];
        ha[r * 288 + 148 + c] = (__bf16)te [(size_t)(g0 + r) * 128 + c];
    }
    for (int i = lane; i < 16 * 12; i += 32) {
        int r = i / 12, c = i % 12;
        ha[r * 288 + 276 + c] = (__bf16)0.0f;         // K padding 276..287
    }

    float ld = 0.0f;                                   // logdet (lanes 0..15, row = lane)

#pragma unroll 1
    for (int l = 0; l < LAYERS; ++l) {
        const __bf16* wl = ws + (size_t)l * LAYER_WS;

        // ---- stage W1T, gather z_a/z_b through composed permutation ----
        stage(sW, wl, W1T_N);
        float za[20], zb[20];
        if (lane < 16) {
            const int r = lane;
#pragma unroll
            for (int j = 0; j < 20; ++j) {
                const int pa = perm[l * 40 + idx_a[l * 20 + j]];
                const int pb = perm[l * 40 + idx_b[l * 20 + j]];
                za[j] = zt[r * 40 + pa];
                zb[j] = zt[r * 40 + pb];
                ha[r * 288 + j] = (__bf16)za[j];
            }
        }
        __syncthreads();

        // ---- h1 = silu([zA|ctx|tE] @ W1 + b1): 16x288 * 288x128 ----
        wave_gemm<288, 288, 128, 9, 8, true, false, true>(ha, sW, b1 + l * 128, hb, nullptr, lane);
        __syncthreads();

        // ---- stage W2T|W3T|WstT as one contiguous block ----
        stage(sW, wl + W2T_OFF, WTAIL_N);
        // prefetch next layer's weight block into L2/L0 behind the coming WMMA stretch
        if (l + 1 < LAYERS) {
            const char* nwl = (const char*)(ws + (size_t)(l + 1) * LAYER_WS);
            for (int i = threadIdx.x; i < (LAYER_WS * 2) / 256; i += 128)
                __builtin_prefetch(nwl + i * 256, 0, 0);
        }
        __syncthreads();

        // ---- h2 = silu(h1 @ W2 + b2); h3 = silu(h2 @ W3 + b3); [s|t] = h3 @ [Ws|Wt] ----
        wave_gemm<128, 128, 128, 4, 8, true, false, true>(hb, sW,         b2 + l * 128, hc, nullptr, lane);
        wave_gemm<128, 128, 128, 4, 8, true, false, true>(hc, sW + 16384, b3 + l * 128, hb, nullptr, lane);
        wave_gemm<128, 128,  64, 4, 4, false, true, false>(hb, sW + 32768, nullptr, nullptr, stw, lane);

        // ---- scalar tail: clip, exp, affine coupling, scatter, logdet ----
        if (lane < 16) {
            const int r = lane;
#pragma unroll
            for (int j = 0; j < 20; ++j) {
                const int iaj = idx_a[l * 20 + j];
                const int ibj = idx_b[l * 20 + j];
                float s = stw[r * 64 + j] + bs[l * 20 + j];
                s = fminf(2.0f, fmaxf(-2.0f, s));
                const float t = stw[r * 64 + 32 + j] + bt[l * 20 + j];
                const float yb = zb[j] * __expf(s) + t;
                ld += s;
                zt[r * 40 + iaj] = za[j];
                zt[r * 40 + ibj] = yb;
            }
        }
        __syncthreads();   // protect sW restage for next layer
    }

    // ---- writeback: z then logdet (outputs concatenated flat) ----
    for (int i = lane; i < 16 * 40; i += 32) {
        int r = i / 40, c = i % 40;
        out[(size_t)(g0 + r) * 40 + c] = zt[r * 40 + c];
    }
    if (lane < 16) out[(size_t)BTOT * 40 + g0 + lane] = ld;
}

extern "C" void kernel_launch(void* const* d_in, const int* in_sizes, int n_in,
                              void* d_out, int out_size, void* d_ws, size_t ws_size,
                              hipStream_t stream) {
    const float* x    = (const float*)d_in[0];
    const float* ctx  = (const float*)d_in[1];
    const float* te   = (const float*)d_in[2];
    const float* W1   = (const float*)d_in[3];
    const float* b1   = (const float*)d_in[4];
    const float* W2   = (const float*)d_in[5];
    const float* b2   = (const float*)d_in[6];
    const float* W3   = (const float*)d_in[7];
    const float* b3   = (const float*)d_in[8];
    const float* Ws   = (const float*)d_in[9];
    const float* bs   = (const float*)d_in[10];
    const float* Wt   = (const float*)d_in[11];
    const float* bt   = (const float*)d_in[12];
    const int*   perm = (const int*)d_in[13];
    const int*   ia   = (const int*)d_in[14];
    const int*   ib   = (const int*)d_in[15];
    __bf16* wsp = (__bf16*)d_ws;
    float*  out = (float*)d_out;

    const int prep_n = LAYERS * LAYER_WS;
    prep_weights<<<(prep_n + 255) / 256, 256, 0, stream>>>(W1, W2, W3, Ws, Wt, wsp);
    flow_kernel<<<BTOT / 64, 128, 0, stream>>>(x, ctx, te, b1, b2, b3, bs, bt,
                                               perm, ia, ib, wsp, out);
}
